// CellInteract_25735444038134
// MI455X (gfx1250) — matrix operand order repaired and checked by reading, not compiled, not run on hardware
//
#include <hip/hip_runtime.h>
#include <hip/hip_bf16.h>

typedef unsigned short u16;
typedef __attribute__((ext_vector_type(16))) __bf16 v16bf;
typedef __attribute__((ext_vector_type(8)))  float  v8f;

union BFrag { v16bf v; uint4 q[2]; };

#define LSCALE_INV2 (1.0f / (100.0f * 100.0f))

// ---------------------------------------------------------------------------
// fp32 -> bf16 (round to nearest even)
// ---------------------------------------------------------------------------
__device__ __forceinline__ u16 f2bf(float x) {
    unsigned int u = __float_as_uint(x);
    unsigned int r = u + 0x7FFFu + ((u >> 16) & 1u);
    return (u16)(r >> 16);
}

// ---------------------------------------------------------------------------
// Liveness fences: force all named fragments into distinct live VGPR blocks at
// this point. Defeats the allocator's one-B-frag-register-recycling pattern,
// so the frag loads batch into clauses with ONE loadcnt drain per WMMA group
// instead of a load->wait->wmma round trip per WMMA.
// ---------------------------------------------------------------------------
__device__ __forceinline__ void fence5(v16bf& a, v16bf& b0, v16bf& b1,
                                       v16bf& b2, v16bf& b3) {
    asm volatile("" : "+v"(a), "+v"(b0), "+v"(b1), "+v"(b2), "+v"(b3));
}
__device__ __forceinline__ void fence3(v16bf& a, v16bf& b0, v16bf& b1) {
    asm volatile("" : "+v"(a), "+v"(b0), "+v"(b1));
}

// ---------------------------------------------------------------------------
// Fragment loaders (wave32), contiguous k order (k = 16h + s) used for both A and B.
// A (16x32 bf16): lane L -> row (L&15); halves[0..15] = K[k0 + 16*(L>>4) .. +16)
// B (32x16 bf16): lane L -> col (L&15); halves[0..15] = K[k0 + 16*(L>>4) .. +16)
// Both loaders read from row-major storage where the K dimension is contiguous
// (for B this means the matrix has been pre-transposed: BT[n][k]).
// ---------------------------------------------------------------------------
__device__ __forceinline__ v16bf load_a_frag(const u16* __restrict__ src, int ld,
                                             int row0, int k0) {
    int lane = threadIdx.x & 31;
    const u16* p = src + (size_t)(row0 + (lane & 15)) * ld + k0 + ((lane >> 4) << 4);
    BFrag f;
    f.q[0] = *(const uint4*)(p);        // K chunk 0 (8 halves, 16B aligned)
    f.q[1] = *(const uint4*)(p + 8);    // K chunk 1
    return f.v;
}

__device__ __forceinline__ v16bf load_b_frag(const u16* __restrict__ srcT, int ld,
                                             int col0, int k0) {
    int lane = threadIdx.x & 31;
    const u16* p = srcT + (size_t)(col0 + (lane & 15)) * ld + k0 + ((lane >> 4) << 4);
    BFrag f;
    f.q[0] = *(const uint4*)(p);        // 16 contiguous halves
    f.q[1] = *(const uint4*)(p + 8);
    return f.v;
}

// ---------------------------------------------------------------------------
// Prep kernels: fp32 -> bf16 convert and transpose+convert (dims are powers of 2)
// ---------------------------------------------------------------------------
__global__ void convert_bf16(const float* __restrict__ in, u16* __restrict__ out, int n) {
    int i = blockIdx.x * blockDim.x + threadIdx.x;
    if (i < n) out[i] = f2bf(in[i]);
}

// in: R x C row-major fp32 ; out: C x R row-major bf16 (out[c*R + r] = in[r*C + c])
// R = 1 << lgR. Linear over output -> coalesced writes.
__global__ void transpose_bf16(const float* __restrict__ in, u16* __restrict__ out,
                               int lgR, int C) {
    int i = blockIdx.x * blockDim.x + threadIdx.x;
    int c = i >> lgR;
    int r = i & ((1 << lgR) - 1);
    out[i] = f2bf(in[(size_t)r * C + c]);
}

// ---------------------------------------------------------------------------
// Generic bf16 WMMA GEMM: C[M,N] = scale * (A[M,K] @ BT[N,K]^T)
// 8 waves/block; each wave owns a 16x64 strip (4 independent accumulation
// chains). Per k-step: batch-load A + 4 B-frags (fence5 -> distinct VGPRs,
// clause'd loads, one drain), then 4 back-to-back WMMAs.
// Output either bf16 (Cbf) or f32 (Cf, non-temporal store: written once).
// ---------------------------------------------------------------------------
__global__ __launch_bounds__(256)
void gemm_bf16_wmma(const u16* __restrict__ A, const u16* __restrict__ BT,
                    u16* __restrict__ Cbf, float* __restrict__ Cf,
                    int N, int K, float scale) {
    int wave = threadIdx.x >> 5;
    int lane = threadIdx.x & 31;
    int job  = blockIdx.x * 8 + wave;
    int ngrp = N >> 6;                       // 64-wide n-groups
    int m0   = (job / ngrp) << 4;
    int n0   = (job % ngrp) << 6;

    v8f acc[4];
#pragma unroll
    for (int t = 0; t < 4; ++t)
#pragma unroll
        for (int i = 0; i < 8; ++i) acc[t][i] = 0.0f;

#pragma unroll 2
    for (int k = 0; k < K; k += 32) {
        // ---- load phase: 5 live frags -> clause'd loads, one drain ----
        v16bf a  = load_a_frag(A, K, m0, k);
        v16bf b0 = load_b_frag(BT, K, n0,      k);
        v16bf b1 = load_b_frag(BT, K, n0 + 16, k);
        v16bf b2 = load_b_frag(BT, K, n0 + 32, k);
        v16bf b3 = load_b_frag(BT, K, n0 + 48, k);
        fence5(a, b0, b1, b2, b3);
        // ---- compute phase: 4 back-to-back WMMAs, independent chains ----
        acc[0] = __builtin_amdgcn_wmma_f32_16x16x32_bf16(false, a, false, b0,
                                                         (short)0, acc[0], false, false);
        acc[1] = __builtin_amdgcn_wmma_f32_16x16x32_bf16(false, a, false, b1,
                                                         (short)0, acc[1], false, false);
        acc[2] = __builtin_amdgcn_wmma_f32_16x16x32_bf16(false, a, false, b2,
                                                         (short)0, acc[2], false, false);
        acc[3] = __builtin_amdgcn_wmma_f32_16x16x32_bf16(false, a, false, b3,
                                                         (short)0, acc[3], false, false);
    }

    int r0 = m0 + ((lane >> 4) << 3);
#pragma unroll
    for (int t = 0; t < 4; ++t) {
        int c = n0 + (t << 4) + (lane & 15);
        if (Cf) {
#pragma unroll
            for (int v = 0; v < 8; ++v)      // streamed once -> TH=NT store
                __builtin_nontemporal_store(acc[t][v] * scale,
                                            &Cf[(size_t)(r0 + v) * N + c]);
        } else {
#pragma unroll
            for (int v = 0; v < 8; ++v)
                Cbf[(size_t)(r0 + v) * N + c] = f2bf(acc[t][v]);
        }
    }
}

// ---------------------------------------------------------------------------
// Fused sigmoid-attention:  out1 = (exp(-d/L^2) * sigmoid(Q @ encB^T)) @ expr
//   Q     : tmpB  [8192 x 256]  bf16 (row-major, K contiguous)
//   encB  : enc   [8192 x 256]  bf16 (row j == column j of enc^T -> B-frag direct)
//   exprT : expr^T[1024 x 8192] bf16 (gene-major so V B-frags are contiguous)
// Block = 16 output rows x 1024 gene cols, 8 waves; wave w owns cols [128w,128w+128).
// Per j-step of 128: each wave builds one gated 16x16 score tile -> LDS P (16x128
// bf16), barrier, then 32 WMMAs of P @ V per wave. V-frags are fenced 4 at a
// time (8 at once would push past ~256 VGPRs with the 64-reg acc file + 64-reg
// Q cache). sqr_pdist (256 MB, streamed once) uses non-temporal loads so it
// cannot evict the L2-resident exprT (16 MB) reused by all 512 blocks.
// ---------------------------------------------------------------------------
__global__ __launch_bounds__(256)
void fused_gated_attn(const u16* __restrict__ Q, const u16* __restrict__ encB,
                      const float* __restrict__ sqr_pdist,
                      const u16* __restrict__ exprT,
                      u16* __restrict__ out1B) {
    __shared__ u16 sP[16 * 128];                     // 4 KB gated-P tile

    const int wave = threadIdx.x >> 5;
    const int lane = threadIdx.x & 31;
    const int row0 = blockIdx.x << 4;
    const int mloc = (lane >> 4) << 3;               // C-layout row base within tile

    // Q fragments for this 16-row block (K = 256 -> 8 frags), held in VGPRs.
    v16bf qf[8];
#pragma unroll
    for (int k = 0; k < 8; ++k) qf[k] = load_a_frag(Q, 256, row0, k * 32);

    // 8 output accumulators: cols wave*128 + t*16
    v8f acc[8];
#pragma unroll
    for (int t = 0; t < 8; ++t)
#pragma unroll
        for (int i = 0; i < 8; ++i) acc[t][i] = 0.0f;

    for (int j = 0; j < 8192; j += 128) {
        // ---- 1) score tile S(16x16) at columns j + wave*16, K = 256 ----
        v8f s;
#pragma unroll
        for (int i = 0; i < 8; ++i) s[i] = 0.0f;
        const int coln = j + (wave << 4);
#pragma unroll
        for (int k = 0; k < 8; k += 2) {             // pairs of live B-frags
            v16bf b0 = load_b_frag(encB, 256, coln, k * 32);
            v16bf b1 = load_b_frag(encB, 256, coln, k * 32 + 32);
            fence3(qf[k], b0, b1);
            s = __builtin_amdgcn_wmma_f32_16x16x32_bf16(false, qf[k], false, b0,
                                                        (short)0, s, false, false);
            s = __builtin_amdgcn_wmma_f32_16x16x32_bf16(false, qf[k + 1], false, b1,
                                                        (short)0, s, false, false);
        }

        // ---- 2) gate: exp(-d/L^2) * sigmoid(s), write bf16 P tile to LDS ----
        const int rr = row0 + mloc;
        const int cc = coln + (lane & 15);
        const float* dp = sqr_pdist + (size_t)rr * 8192 + cc;
        if (j + 128 < 8192)                          // gfx1250 global_prefetch_b8
            __builtin_prefetch(dp + 128, 0, 1);
#pragma unroll
        for (int v = 0; v < 8; ++v) {
            float d   = __builtin_nontemporal_load(&dp[(size_t)v * 8192]);
            float sig = 1.0f / (1.0f + __expf(-s[v]));
            float g   = __expf(-d * LSCALE_INV2) * sig;
            sP[(mloc + v) * 128 + (wave << 4) + (lane & 15)] = f2bf(g);
        }
        __syncthreads();

        // ---- 3) O += P(16x128) @ V(128 x 128-col-slice) ----
        const int vcol = wave << 7;
#pragma unroll
        for (int kk = 0; kk < 4; ++kk) {
            v16bf pf = load_a_frag(sP, 128, 0, kk * 32);     // from LDS (ds_load_b128)
            const int kg = j + kk * 32;
            // group 0: 4 fenced live V-frags -> one drain, 4 WMMAs
            {
                v16bf v0 = load_b_frag(exprT, 8192, vcol,      kg);
                v16bf v1 = load_b_frag(exprT, 8192, vcol + 16, kg);
                v16bf v2 = load_b_frag(exprT, 8192, vcol + 32, kg);
                v16bf v3 = load_b_frag(exprT, 8192, vcol + 48, kg);
                fence5(pf, v0, v1, v2, v3);
                acc[0] = __builtin_amdgcn_wmma_f32_16x16x32_bf16(false, pf, false, v0,
                                                                 (short)0, acc[0], false, false);
                acc[1] = __builtin_amdgcn_wmma_f32_16x16x32_bf16(false, pf, false, v1,
                                                                 (short)0, acc[1], false, false);
                acc[2] = __builtin_amdgcn_wmma_f32_16x16x32_bf16(false, pf, false, v2,
                                                                 (short)0, acc[2], false, false);
                acc[3] = __builtin_amdgcn_wmma_f32_16x16x32_bf16(false, pf, false, v3,
                                                                 (short)0, acc[3], false, false);
            }
            // group 1: 4 fenced live V-frags
            {
                v16bf v4 = load_b_frag(exprT, 8192, vcol + 64,  kg);
                v16bf v5 = load_b_frag(exprT, 8192, vcol + 80,  kg);
                v16bf v6 = load_b_frag(exprT, 8192, vcol + 96,  kg);
                v16bf v7 = load_b_frag(exprT, 8192, vcol + 112, kg);
                fence5(pf, v4, v5, v6, v7);
                acc[4] = __builtin_amdgcn_wmma_f32_16x16x32_bf16(false, pf, false, v4,
                                                                 (short)0, acc[4], false, false);
                acc[5] = __builtin_amdgcn_wmma_f32_16x16x32_bf16(false, pf, false, v5,
                                                                 (short)0, acc[5], false, false);
                acc[6] = __builtin_amdgcn_wmma_f32_16x16x32_bf16(false, pf, false, v6,
                                                                 (short)0, acc[6], false, false);
                acc[7] = __builtin_amdgcn_wmma_f32_16x16x32_bf16(false, pf, false, v7,
                                                                 (short)0, acc[7], false, false);
            }
        }
        __syncthreads();                             // protect sP for next iteration
    }

    // ---- store out1 as bf16 (input of the final GEMM) ----
#pragma unroll
    for (int t = 0; t < 8; ++t) {
        int c = (wave << 7) + (t << 4) + (lane & 15);
#pragma unroll
        for (int v = 0; v < 8; ++v)
            out1B[(size_t)(row0 + mloc + v) * 1024 + c] = f2bf(acc[t][v]);
    }
}

// ---------------------------------------------------------------------------
// Host-side launch (graph-capture safe: only kernel launches on `stream`)
// ---------------------------------------------------------------------------
extern "C" void kernel_launch(void* const* d_in, const int* in_sizes, int n_in,
                              void* d_out, int out_size, void* d_ws, size_t ws_size,
                              hipStream_t stream) {
    const float* expression    = (const float*)d_in[0];  // [8192,1024]
    const float* encoding      = (const float*)d_in[1];  // [8192, 256]
    const float* sqr_pdist     = (const float*)d_in[2];  // [8192,8192]
    const float* transform     = (const float*)d_in[3];  // [ 256, 256]
    const float* gene_response = (const float*)d_in[4];  // [1024,1024]
    float* out = (float*)d_out;                          // [8192,1024] f32

    const int N = 8192, DE = 256, DG = 1024;

    // workspace layout (bf16, ~42 MB total)
    u16* encB  = (u16*)d_ws;                 // [8192 x 256]
    u16* tfT   = encB  + (size_t)N * DE;     // [ 256 x 256]  transform^T
    u16* exprT = tfT   + (size_t)DE * DE;    // [1024 x 8192] expression^T
    u16* grT   = exprT + (size_t)DG * N;     // [1024 x 1024] gene_response^T
    u16* tmpB  = grT   + (size_t)DG * DG;    // [8192 x 256]  Q = enc @ transform
    u16* out1B = tmpB  + (size_t)N * DE;     // [8192 x 1024] gated @ expression

    // 1) precision conversion / transposes (lgR = log2(rows of source))
    convert_bf16<<<(N * DE) / 256, 256, 0, stream>>>(encoding, encB, N * DE);
    transpose_bf16<<<(DE * DE) / 256, 256, 0, stream>>>(transform, tfT, 8, DE);
    transpose_bf16<<<(N * DG) / 256, 256, 0, stream>>>(expression, exprT, 13, DG);
    transpose_bf16<<<(DG * DG) / 256, 256, 0, stream>>>(gene_response, grT, 10, DG);

    // 2) Q = encoding @ transform   (8192x256x256, bf16 out)
    //    jobs = (8192/16)*(256/64) = 2048; 8 waves/block -> 256 blocks
    gemm_bf16_wmma<<<256, 256, 0, stream>>>(encB, tfT, tmpB, nullptr, DE, DE, 1.0f);

    // 3) out1 = (exp(-d/L^2) * sigmoid(Q @ enc^T)) @ expression   (fused, no NxN tensor)
    fused_gated_attn<<<N / 16, 256, 0, stream>>>(tmpB, encB, sqr_pdist, exprT, out1B);

    // 4) out = out1 @ gene_response / 1024   (8192x1024x1024, f32 out)
    //    jobs = (8192/16)*(1024/64) = 8192 -> 1024 blocks
    gemm_bf16_wmma<<<1024, 256, 0, stream>>>(out1B, grT, nullptr, out, DG, DG,
                                             1.0f / (float)DG);
}
